// GlowTTSMultiHeadAttention_85504208928804
// MI455X (gfx1250) — matrix operand order repaired and checked by reading, not compiled, not run on hardware
//
#include <hip/hip_runtime.h>

#define BB 4
#define CC 1024
#define TTT 2048
#define NH 16
#define DH 64

typedef __attribute__((ext_vector_type(16))) __bf16 v16bf;
typedef __attribute__((ext_vector_type(8)))  float  v8f;
typedef __attribute__((ext_vector_type(4)))  float  f32x4;
typedef __attribute__((ext_vector_type(4)))  unsigned int u32x4;
typedef __attribute__((ext_vector_type(4)))  __bf16 bf16x4;

__device__ __forceinline__ v16bf load_frag16(const __bf16* p0, const __bf16* p1) {
  union { v16bf v; u32x4 q[2]; } u;
  u.q[0] = *(const u32x4*)p0;   // 16B (8 bf16) chunk
  u.q[1] = *(const u32x4*)p1;   // 16B (8 bf16) chunk
  return u.v;
}

// ---------------------------------------------------------------------------
// f32 -> bf16 conversion (vectorized)
// ---------------------------------------------------------------------------
__global__ __launch_bounds__(256)
void cvt_kernel(const float* __restrict__ src, __bf16* __restrict__ dst, int n) {
  int i = (blockIdx.x * 256 + threadIdx.x) * 4;
  if (i < n) {
    f32x4 v = *(const f32x4*)(src + i);
    bf16x4 o;
    o[0] = (__bf16)v[0]; o[1] = (__bf16)v[1];
    o[2] = (__bf16)v[2]; o[3] = (__bf16)v[3];
    *(bf16x4*)(dst + i) = o;
  }
}

// ---------------------------------------------------------------------------
// out[o,t] = sum_c W[o,c] * X[b,c,t] + bias[o], with mode-specific epilogue
//   mode 0: RoPE(D_ROPE=32) + store bf16 [b,h,t,d]   (q and k)
//   mode 1: store bf16 [b,h,d,t]                     (v)
//   mode 2: store f32  [b,o,t]                       (final projection)
// Tile: 128(M) x 128(N) per workgroup, 8 waves of 64x32, K staged via LDS.
// ---------------------------------------------------------------------------
__global__ __launch_bounds__(256)
void gemm_kernel(const __bf16* __restrict__ W,     // [1024 x 1024] row-major
                 const __bf16* __restrict__ X,     // [B][C][T] bf16
                 const float*  __restrict__ bias,  // [1024]
                 __bf16* __restrict__ out16,
                 float*  __restrict__ outf,
                 int mode)
{
  __shared__ __bf16 Al[128 * 40];   // [m][k], k-stride 40 (pad 8)
  __shared__ __bf16 Bl[128 * 40];   // [n][k] (transposed), k-stride 40

  const int tid   = threadIdx.x;
  const int b     = blockIdx.z;
  const int n0    = blockIdx.x * 128;
  const int m0    = blockIdx.y * 128;
  const __bf16* Xb = X + (size_t)b * CC * TTT;

  const int lane    = tid & 31;
  const int wave    = tid >> 5;
  const int waveM   = wave >> 2;          // 0..1
  const int waveN   = wave & 3;           // 0..3
  const int lm      = lane & 15;
  const int halfsel = (lane < 16) ? 0 : 1;
  const int aoff    = halfsel * 8;
  const int boff    = halfsel * 16;

  v8f acc[4][2];
#pragma unroll
  for (int mi = 0; mi < 4; ++mi)
#pragma unroll
    for (int ni = 0; ni < 2; ++ni)
      acc[mi][ni] = (v8f)(0.0f);

  for (int k0 = 0; k0 < CC; k0 += 32) {
    // --- stage A tile (128x32, row-major) ---
#pragma unroll
    for (int i = 0; i < 2; ++i) {
      int idx = tid + i * 256;            // 0..511 chunks of 8
      int row = idx >> 2;
      int cc  = (idx & 3) * 8;
      u32x4 d = *(const u32x4*)(W + (size_t)(m0 + row) * CC + k0 + cc);
      *(u32x4*)(&Al[row * 40 + cc]) = d;
    }
    // --- stage B tile transposed: Bl[t][k] from X[k][t] ---
#pragma unroll
    for (int i = 0; i < 2; ++i) {
      int idx = tid + i * 256;
      int kk  = idx >> 4;                 // 0..31
      int ttc = (idx & 15) * 8;           // 0..120
      union { u32x4 q; __bf16 h[8]; } d;
      d.q = *(const u32x4*)(Xb + (size_t)(k0 + kk) * TTT + n0 + ttc);
#pragma unroll
      for (int j = 0; j < 8; ++j)
        Bl[(ttc + j) * 40 + kk] = d.h[j];
    }
    __syncthreads();

    v16bf af[4], bf[2];
#pragma unroll
    for (int mi = 0; mi < 4; ++mi) {
      const __bf16* p = &Al[(waveM * 64 + mi * 16 + lm) * 40 + aoff];
      af[mi] = load_frag16(p, p + 16);    // K = {aoff..aoff+7, 16+aoff..}
    }
#pragma unroll
    for (int ni = 0; ni < 2; ++ni) {
      const __bf16* p = &Bl[(waveN * 32 + ni * 16 + lm) * 40 + aoff];
      bf[ni] = load_frag16(p, p + 16);    // K = {aoff..aoff+7, 16+aoff..}
    }
#pragma unroll
    for (int mi = 0; mi < 4; ++mi)
#pragma unroll
      for (int ni = 0; ni < 2; ++ni)
        acc[mi][ni] = __builtin_amdgcn_wmma_f32_16x16x32_bf16(
            false, af[mi], false, bf[ni], (short)0, acc[mi][ni], false, false);
    __syncthreads();
  }

  // ---- epilogue ----
  const int head = (m0 + waveM * 64) >> 6;

#pragma unroll
  for (int mi = 0; mi < 4; ++mi)
#pragma unroll
    for (int r = 0; r < 8; ++r) {
      int orow = m0 + waveM * 64 + mi * 16 + r + halfsel * 8;
      float bv = bias[orow];
      acc[mi][0][r] += bv;
      acc[mi][1][r] += bv;
    }

  if (mode == 0) {
    // RoPE on d in [0,32): tiles mi=0 (d=i) and mi=1 (d=16+i) pair up.
#pragma unroll
    for (int ni = 0; ni < 2; ++ni) {
      int t = n0 + waveN * 32 + ni * 16 + lm;
#pragma unroll
      for (int r = 0; r < 8; ++r) {
        int i = r + halfsel * 8;                       // 0..15
        float theta = __powf(10000.0f, -(float)i * (1.0f / 16.0f));
        float ang = (float)t * theta;
        float cs = __cosf(ang), sn = __sinf(ang);
        float v0 = acc[0][ni][r], v1 = acc[1][ni][r];
        acc[0][ni][r] = v0 * cs - v1 * sn;
        acc[1][ni][r] = v1 * cs + v0 * sn;
      }
    }
  }

#pragma unroll
  for (int mi = 0; mi < 4; ++mi)
#pragma unroll
    for (int ni = 0; ni < 2; ++ni)
#pragma unroll
      for (int r = 0; r < 8; ++r) {
        int d = mi * 16 + r + halfsel * 8;             // 0..63 within head
        int t = n0 + waveN * 32 + ni * 16 + lm;
        float v = acc[mi][ni][r];
        if (mode == 0) {
          out16[(((size_t)b * NH + head) * TTT + t) * DH + d] = (__bf16)v;
        } else if (mode == 1) {
          out16[(((size_t)b * NH + head) * DH + d) * TTT + t] = (__bf16)v;
        } else {
          int orow = m0 + waveM * 64 + mi * 16 + r + halfsel * 8;
          outf[((size_t)b * CC + orow) * TTT + t] = v;
        }
      }
}

// ---------------------------------------------------------------------------
// Flash attention: one wave = 16 query rows, stream keys in chunks of 32.
// q,k: bf16 [b,h,t,64]; v: bf16 [b,h,64,t]; out: bf16 [b, h*64+d, t].
// ---------------------------------------------------------------------------
__global__ __launch_bounds__(128)
void attn_kernel(const __bf16* __restrict__ q,
                 const __bf16* __restrict__ k,
                 const __bf16* __restrict__ v,
                 const float*  __restrict__ mask,   // [B,1,T,T]
                 __bf16* __restrict__ att)
{
  __shared__ __bf16 Pl[4 * 16 * 40];   // per-wave 16x32 P tile, k-stride 40

  const int tid     = threadIdx.x;
  const int lane    = tid & 31;
  const int wave    = tid >> 5;
  const int lm      = lane & 15;
  const int halfsel = (lane < 16) ? 0 : 1;
  const int aoff    = halfsel * 8;

  const int b   = blockIdx.z;
  const int h   = blockIdx.y;
  const int bh  = b * NH + h;
  const int tq0 = blockIdx.x * 64 + wave * 16;

  const __bf16* qb = q + (size_t)bh * TTT * DH;
  const __bf16* kb = k + (size_t)bh * TTT * DH;
  const __bf16* vb = v + (size_t)bh * DH * TTT;

  // preload Q fragments (16 rows x 64 d = 2 K-chunks)
  v16bf qf[2];
#pragma unroll
  for (int s = 0; s < 2; ++s) {
    const __bf16* p = qb + (size_t)(tq0 + lm) * DH + s * 32 + aoff;
    qf[s] = load_frag16(p, p + 16);
  }

  v8f O[4];
#pragma unroll
  for (int j = 0; j < 4; ++j) O[j] = (v8f)(0.0f);
  float rowm[8], rowl[8];
#pragma unroll
  for (int r = 0; r < 8; ++r) { rowm[r] = -3.0e38f; rowl[r] = 0.0f; }

  for (int tk0 = 0; tk0 < TTT; tk0 += 32) {
    // --- S = q @ k^T for 16 x 32 keys (2 n-tiles x 2 k-steps) ---
    v8f S[2];
#pragma unroll
    for (int j = 0; j < 2; ++j) S[j] = (v8f)(0.0f);
#pragma unroll
    for (int j = 0; j < 2; ++j)
#pragma unroll
      for (int s = 0; s < 2; ++s) {
        const __bf16* p = kb + (size_t)(tk0 + j * 16 + lm) * DH + s * 32 + halfsel * 8;
        v16bf kf = load_frag16(p, p + 16);
        S[j] = __builtin_amdgcn_wmma_f32_16x16x32_bf16(
            false, qf[s], false, kf, (short)0, S[j], false, false);
      }

    // --- online softmax (scale 1/sqrt(64), mask, exp, rescale O) ---
#pragma unroll
    for (int r = 0; r < 8; ++r) {
      int tq = tq0 + r + halfsel * 8;
      const float* mrow = mask + ((size_t)b * TTT + tq) * TTT + tk0;
      float mv0 = mrow[lm];
      float mv1 = mrow[16 + lm];
      float s0 = S[0][r] * 0.125f;
      float s1 = S[1][r] * 0.125f;
      s0 = (mv0 == 0.0f) ? -10000.0f : s0;
      s1 = (mv1 == 0.0f) ? -10000.0f : s1;
      float cmax = fmaxf(s0, s1);
#pragma unroll
      for (int m = 8; m >= 1; m >>= 1)
        cmax = fmaxf(cmax, __shfl_xor(cmax, m, 32));
      float mnew = fmaxf(rowm[r], cmax);
      float corr = __expf(rowm[r] - mnew);
      rowm[r] = mnew;
      float p0 = __expf(s0 - mnew);
      float p1 = __expf(s1 - mnew);
      rowl[r] = rowl[r] * corr + p0 + p1;   // per-lane partial; reduce at end
#pragma unroll
      for (int j = 0; j < 4; ++j) O[j][r] *= corr;
      int mr = r + halfsel * 8;
      Pl[wave * 640 + mr * 40 + lm]      = (__bf16)p0;
      Pl[wave * 640 + mr * 40 + 16 + lm] = (__bf16)p1;
    }
    asm volatile("s_wait_dscnt 0" ::: "memory");   // C-layout -> A-frag handoff

    // --- P fragment (16 x 32) from LDS ---
    const __bf16* pp = &Pl[wave * 640 + lm * 40 + aoff];
    v16bf pf = load_frag16(pp, pp + 16);

    // --- O += P @ V ---
#pragma unroll
    for (int j = 0; j < 4; ++j) {
      const __bf16* p = vb + (size_t)(j * 16 + lm) * TTT + tk0 + halfsel * 8;
      v16bf vf = load_frag16(p, p + 16);
      O[j] = __builtin_amdgcn_wmma_f32_16x16x32_bf16(
          false, pf, false, vf, (short)0, O[j], false, false);
    }
  }

  // --- normalize and store att[b, h*64+d, tq] (c-major for final GEMM) ---
#pragma unroll
  for (int r = 0; r < 8; ++r) {
    float l = rowl[r];
#pragma unroll
    for (int m = 8; m >= 1; m >>= 1)
      l += __shfl_xor(l, m, 32);
    float rinv = 1.0f / l;
    int tq = tq0 + r + halfsel * 8;
#pragma unroll
    for (int j = 0; j < 4; ++j) {
      int d = j * 16 + lm;
      att[((size_t)bh * DH + d) * TTT + tq] = (__bf16)(O[j][r] * rinv);
    }
  }
}

// ---------------------------------------------------------------------------
extern "C" void kernel_launch(void* const* d_in, const int* in_sizes, int n_in,
                              void* d_out, int out_size, void* d_ws, size_t ws_size,
                              hipStream_t stream) {
  (void)in_sizes; (void)n_in; (void)out_size; (void)ws_size;
  const float* x    = (const float*)d_in[0];
  const float* c    = (const float*)d_in[1];
  const float* mask = (const float*)d_in[2];
  const float* Wq   = (const float*)d_in[3];
  const float* bq   = (const float*)d_in[4];
  const float* Wk   = (const float*)d_in[5];
  const float* bk   = (const float*)d_in[6];
  const float* Wv   = (const float*)d_in[7];
  const float* bv   = (const float*)d_in[8];
  const float* Wo   = (const float*)d_in[9];
  const float* bo   = (const float*)d_in[10];

  char* ws = (char*)d_ws;
  size_t off = 0;
  auto take = [&](size_t nelem) {
    __bf16* p = (__bf16*)(ws + off);
    off += nelem * sizeof(__bf16);
    return p;
  };
  const size_t WN = (size_t)CC * CC;          // 1M
  const size_t XN = (size_t)BB * CC * TTT;    // 8M
  __bf16* Wq16 = take(WN);
  __bf16* Wk16 = take(WN);
  __bf16* Wv16 = take(WN);
  __bf16* Wo16 = take(WN);
  __bf16* x16  = take(XN);
  __bf16* c16  = take(XN);
  __bf16* q16  = take(XN);   // [b,h,t,d]
  __bf16* k16  = take(XN);   // [b,h,t,d]
  __bf16* v16  = take(XN);   // [b,h,d,t]
  __bf16* a16  = take(XN);   // [b, h*64+d, t]

  auto cv = [&](const float* s, __bf16* d, size_t n) {
    unsigned g = (unsigned)((n / 4 + 255) / 256);
    cvt_kernel<<<dim3(g), dim3(256), 0, stream>>>(s, d, (int)n);
  };
  cv(Wq, Wq16, WN);
  cv(Wk, Wk16, WN);
  cv(Wv, Wv16, WN);
  cv(Wo, Wo16, WN);
  cv(x,  x16,  XN);
  cv(c,  c16,  XN);

  dim3 gg(TTT / 128, CC / 128, BB), gb(256);
  gemm_kernel<<<gg, gb, 0, stream>>>(Wq16, x16, bq, q16, nullptr, 0);
  gemm_kernel<<<gg, gb, 0, stream>>>(Wk16, c16, bk, k16, nullptr, 0);
  gemm_kernel<<<gg, gb, 0, stream>>>(Wv16, c16, bv, v16, nullptr, 1);

  attn_kernel<<<dim3(TTT / 64, NH, BB), dim3(128), 0, stream>>>(q16, k16, v16, mask, a16);

  gemm_kernel<<<gg, gb, 0, stream>>>(Wo16, a16, bo, nullptr, (float*)d_out, 2);
}
